// NonLinearReadoutBlock_28939489641116
// MI455X (gfx1250) — matrix operand order repaired and checked by reading, not compiled, not run on hardware
//
#include <hip/hip_runtime.h>
#include <math.h>

// ---- WMMA vector types (per CDNA5_HIP.md) ----
typedef __attribute__((ext_vector_type(16))) _Float16 v16h;
typedef __attribute__((ext_vector_type(8)))  _Float16 v8h;
typedef __attribute__((ext_vector_type(8)))  float    v8f;
typedef __attribute__((ext_vector_type(4)))  float    v4f;

#define D_IN   256
#define D_HID  128
#define D_OUT  4
#define NT     8                    // n-tiles of 16 across D_HID
#define KT     8                    // k-steps of 32 across D_IN
#define KT2    4                    // k-steps of 32 across D_HID
#define W1FRAG_HALFS (NT * KT * 32 * 16)   // 32768 halfs = 64 KB
#define W2FRAG_HALFS (KT2 * 32 * 16)       // 2048 halfs  = 4 KB
#define WAVES  8
#define MAX_BLOCKS 1024

// Fragment layouts (wave32):
//   A (16x32 f16): lane = 16*half + m (m=row).  Element e of v16h holds
//     K = 32*kt + 8*half + e          (e < 8)
//     K = 32*kt + 8*half + 16 + (e-8) (e >= 8)
//   B (32x16 f16): lane = 16*half + n (n=col).  Element e holds
//     K = same mapping as A (8*half + e; +16 for e >= 8)
//   C/D (16x16 f32): lane = 16*half + n; VGPR c holds row m = c + 8*half.

// ---------- prepack: W1/W2 fp32 -> f16 B-fragments in d_ws ----------
__global__ void nlrb_prepack(const float* __restrict__ W1,
                             const float* __restrict__ W2,
                             _Float16* __restrict__ wsW1,
                             _Float16* __restrict__ wsW2) {
  int t = blockIdx.x * blockDim.x + threadIdx.x;
  if (t < W1FRAG_HALFS) {
    int e    = t & 15;
    int lane = (t >> 4) & 31;
    int kt   = (t >> 9) & 7;
    int nt   = (t >> 12) & 7;
    int n    = nt * 16 + (lane & 15);
    int K    = kt * 32 + (lane >> 4) * 8 + (e & 7) + (e >> 3) * 16;   // B-fragment K mapping
    wsW1[t]  = (_Float16)W1[K * D_HID + n];
  } else if (t < W1FRAG_HALFS + W2FRAG_HALFS) {
    int t2   = t - W1FRAG_HALFS;
    int e    = t2 & 15;
    int lane = (t2 >> 4) & 31;
    int kt   = t2 >> 9;
    int n    = lane & 15;
    int K    = kt * 32 + (lane >> 4) * 8 + (e & 7) + (e >> 3) * 16;
    wsW2[t2] = (n < D_OUT) ? (_Float16)W2[K * D_OUT + n] : (_Float16)0.0f;
  }
}

// Load all 8 B-fragments of one k-step from LDS with explicit ds_load_b128 +
// one s_wait_dscnt. Inline asm keeps these loads in-loop (no hoisting into
// 600 VGPRs, no scratch spills, no volatile->flat/SCOPE_SYS mangling).
// addr = lds_byte_addr(sW1) + lane*32 + kt*1024; fragment nt at +nt*8192.
struct BRegs { v8h r0,r1,r2,r3,r4,r5,r6,r7,r8,r9,r10,r11,r12,r13,r14,r15; };

__device__ __forceinline__ void load_b_fragments(unsigned addr, BRegs& b) {
  asm volatile(
      "ds_load_b128 %0,  %16\n\t"
      "ds_load_b128 %1,  %16 offset:16\n\t"
      "ds_load_b128 %2,  %16 offset:8192\n\t"
      "ds_load_b128 %3,  %16 offset:8208\n\t"
      "ds_load_b128 %4,  %16 offset:16384\n\t"
      "ds_load_b128 %5,  %16 offset:16400\n\t"
      "ds_load_b128 %6,  %16 offset:24576\n\t"
      "ds_load_b128 %7,  %16 offset:24592\n\t"
      "ds_load_b128 %8,  %16 offset:32768\n\t"
      "ds_load_b128 %9,  %16 offset:32784\n\t"
      "ds_load_b128 %10, %16 offset:40960\n\t"
      "ds_load_b128 %11, %16 offset:40976\n\t"
      "ds_load_b128 %12, %16 offset:49152\n\t"
      "ds_load_b128 %13, %16 offset:49168\n\t"
      "ds_load_b128 %14, %16 offset:57344\n\t"
      "ds_load_b128 %15, %16 offset:57360\n\t"
      "s_wait_dscnt 0x0"
      : "=v"(b.r0), "=v"(b.r1), "=v"(b.r2),  "=v"(b.r3),
        "=v"(b.r4), "=v"(b.r5), "=v"(b.r6),  "=v"(b.r7),
        "=v"(b.r8), "=v"(b.r9), "=v"(b.r10), "=v"(b.r11),
        "=v"(b.r12),"=v"(b.r13),"=v"(b.r14), "=v"(b.r15)
      : "v"(addr));
}

#define CAT16(lo, hi) __builtin_shufflevector(lo, hi, \
    0, 1, 2, 3, 4, 5, 6, 7, 8, 9, 10, 11, 12, 13, 14, 15)

// ---------- fused readout: x@W1 -> silu -> head-mask -> @W2 ----------
__global__ __launch_bounds__(WAVES * 32)
void nlrb_readout(const float* __restrict__ x,
                  const int*   __restrict__ heads,
                  const _Float16* __restrict__ wsW1,
                  const _Float16* __restrict__ wsW2,
                  float* __restrict__ out, int N) {
  __shared__ __attribute__((aligned(32))) _Float16 sW1[W1FRAG_HALFS];      // 64 KB
  __shared__ __attribute__((aligned(32))) _Float16 sH[WAVES * 16 * D_HID]; // 32 KB

  // cooperative copy of W1 fragments into LDS (16B chunks)
  {
    const uint4* src = (const uint4*)wsW1;
    uint4*       dst = (uint4*)sW1;
    for (int i = threadIdx.x; i < W1FRAG_HALFS / 8; i += blockDim.x)
      dst[i] = src[i];
  }
  __syncthreads();

  const int lane = threadIdx.x & 31;
  const int wave = threadIdx.x >> 5;
  const int m    = lane & 15;     // row (A) or col (B/C) within tile
  const int half = lane >> 4;

  _Float16* hstage = sH + wave * (16 * D_HID);
  // 32-bit LDS byte address of this lane's B-fragment base
  const unsigned bbase = (unsigned)(uintptr_t)(sW1) + (unsigned)lane * 32u;

  // W2 fragments: 4 x v16h = 32 VGPRs, loop-invariant -> keep in registers.
  v16h w2frag[KT2];
#pragma unroll
  for (int kt = 0; kt < KT2; ++kt)
    w2frag[kt] = *(const v16h*)(wsW2 + ((size_t)(kt * 32 + lane)) * 16);

  const int totalWaves = (int)gridDim.x * WAVES;
  const int gwave      = (int)blockIdx.x * WAVES + wave;
  const int tiles      = (N + 15) >> 4;

  for (int tile = gwave; tile < tiles; tile += totalWaves) {
    const int r0 = tile * 16;
    int rowm = r0 + m; if (rowm > N - 1) rowm = N - 1;
    const float* xrow = x + (size_t)rowm * D_IN;

    // prefetch next tile's row of x while we compute this one
    {
      long nt_row = (long)(tile + totalWaves) * 16 + m;
      if (nt_row < N)
        __builtin_prefetch(x + (size_t)nt_row * D_IN, 0, 3);
    }

    // per-lane head ids for the 8 rows this lane's C-VGPRs cover
    int hd[8];
#pragma unroll
    for (int c = 0; c < 8; ++c) {
      int r = r0 + c + 8 * half;
      hd[c] = heads[r < N ? r : N - 1];
    }

    // ---- preload & convert ALL A-fragments for this tile (no asm in this
    // loop: the compiler can clause all 32 global_load_b128 with counted
    // waits and overlap the fp32->f16 converts) ----
    v16h afrag[KT];
#pragma unroll
    for (int kt = 0; kt < KT; ++kt) {
      const float* p0 = xrow + kt * 32 + half * 8;
      v4f f0 = *(const v4f*)(p0);
      v4f f1 = *(const v4f*)(p0 + 4);
      v4f f2 = *(const v4f*)(p0 + 16);
      v4f f3 = *(const v4f*)(p0 + 20);
      v16h a;
#pragma unroll
      for (int j = 0; j < 4; ++j) {
        a[j]      = (_Float16)f0[j];
        a[4 + j]  = (_Float16)f1[j];
        a[8 + j]  = (_Float16)f2[j];
        a[12 + j] = (_Float16)f3[j];
      }
      afrag[kt] = a;
    }

    v8f acc[NT];
#pragma unroll
    for (int nt = 0; nt < NT; ++nt) { v8f z = {}; acc[nt] = z; }

    // ---- main WMMA loop: per k-step, one ds_load burst then 8 WMMAs ----
#pragma unroll
    for (int kt = 0; kt < KT; ++kt) {
      BRegs br;
      load_b_fragments(bbase + (unsigned)kt * 1024u, br);
      acc[0] = __builtin_amdgcn_wmma_f32_16x16x32_f16(
          false, afrag[kt], false, CAT16(br.r0,  br.r1),  (short)0, acc[0], false, false);
      acc[1] = __builtin_amdgcn_wmma_f32_16x16x32_f16(
          false, afrag[kt], false, CAT16(br.r2,  br.r3),  (short)0, acc[1], false, false);
      acc[2] = __builtin_amdgcn_wmma_f32_16x16x32_f16(
          false, afrag[kt], false, CAT16(br.r4,  br.r5),  (short)0, acc[2], false, false);
      acc[3] = __builtin_amdgcn_wmma_f32_16x16x32_f16(
          false, afrag[kt], false, CAT16(br.r6,  br.r7),  (short)0, acc[3], false, false);
      acc[4] = __builtin_amdgcn_wmma_f32_16x16x32_f16(
          false, afrag[kt], false, CAT16(br.r8,  br.r9),  (short)0, acc[4], false, false);
      acc[5] = __builtin_amdgcn_wmma_f32_16x16x32_f16(
          false, afrag[kt], false, CAT16(br.r10, br.r11), (short)0, acc[5], false, false);
      acc[6] = __builtin_amdgcn_wmma_f32_16x16x32_f16(
          false, afrag[kt], false, CAT16(br.r12, br.r13), (short)0, acc[6], false, false);
      acc[7] = __builtin_amdgcn_wmma_f32_16x16x32_f16(
          false, afrag[kt], false, CAT16(br.r14, br.r15), (short)0, acc[7], false, false);
    }

    // epilogue: scale (1/sqrt(256)), silu, head mask; stage f16 h-tile in LDS.
    // channel-block of n-tile nt is nt>>1 (32 channels per head block).
#pragma unroll
    for (int nt = 0; nt < NT; ++nt) {
      const int blk = nt >> 1;
#pragma unroll
      for (int c = 0; c < 8; ++c) {
        float v = acc[nt][c] * 0.0625f;
        v = v / (1.0f + __expf(-v));            // silu
        v = (hd[c] == blk) ? v : 0.0f;
        hstage[(c + 8 * half) * D_HID + nt * 16 + m] = (_Float16)v;
      }
    }

    // second matmul: h[16x128] @ W2pad[128x16] via 4 WMMAs (A from LDS)
    v8f acc2 = {};
#pragma unroll
    for (int kt = 0; kt < KT2; ++kt) {
      const _Float16* q = hstage + m * D_HID + kt * 32 + half * 8;
      v8h lo = *(const v8h*)(q);
      v8h hi = *(const v8h*)(q + 16);
      v16h a2 = CAT16(lo, hi);
      acc2 = __builtin_amdgcn_wmma_f32_16x16x32_f16(
          false, a2, false, w2frag[kt], (short)0, acc2, false, false);
    }

    // store valid 4 columns, scaled by 1/sqrt(128)
    if (m < D_OUT) {
#pragma unroll
      for (int c = 0; c < 8; ++c) {
        int r = r0 + c + 8 * half;
        if (r < N)
          out[(size_t)r * D_OUT + m] = acc2[c] * 0.08838834764831845f;
      }
    }
  }
}

extern "C" void kernel_launch(void* const* d_in, const int* in_sizes, int n_in,
                              void* d_out, int out_size, void* d_ws, size_t ws_size,
                              hipStream_t stream) {
  const float* x     = (const float*)d_in[0];
  const float* W1    = (const float*)d_in[1];
  const float* W2    = (const float*)d_in[2];
  const int*   heads = (const int*)d_in[3];
  float*       out   = (float*)d_out;
  const int N = in_sizes[3];

  _Float16* wsW1 = (_Float16*)d_ws;
  _Float16* wsW2 = wsW1 + W1FRAG_HALFS;

  {
    const int total = W1FRAG_HALFS + W2FRAG_HALFS;
    nlrb_prepack<<<(total + 255) / 256, 256, 0, stream>>>(W1, W2, wsW1, wsW2);
  }

  const int tiles  = (N + 15) / 16;
  int blocks = (tiles + WAVES - 1) / WAVES;
  if (blocks > MAX_BLOCKS) blocks = MAX_BLOCKS;
  if (blocks < 1) blocks = 1;
  nlrb_readout<<<blocks, WAVES * 32, 0, stream>>>(x, heads, wsW1, wsW2, out, N);
}
